// KnowledgeGraphNN_87900800680715
// MI455X (gfx1250) — matrix operand order repaired and checked by reading, not compiled, not run on hardware
//
#include <hip/hip_runtime.h>

#define N_NODES 4096
#define DIM     384
#define NE      131072
#define NH      8
#define DH      48
#define NL      3

typedef __attribute__((ext_vector_type(16))) __bf16 v16bf;
typedef __attribute__((ext_vector_type(8)))  __bf16 v8bf;
typedef __attribute__((ext_vector_type(8)))  float  v8f;

__device__ __forceinline__ unsigned short f2bfbits(float f) {
    unsigned u = __builtin_bit_cast(unsigned, f);
    unsigned r = u + 0x7FFFu + ((u >> 16) & 1u);   // round-to-nearest-even
    return (unsigned short)(r >> 16);
}
__device__ __forceinline__ __bf16 bits2bf(unsigned short s) {
    return __builtin_bit_cast(__bf16, s);
}

#ifdef __has_builtin
#if __has_builtin(__builtin_amdgcn_cvt_pk_bf16_f32)
#define HAVE_PK_BF16 1
#endif
#endif

#ifdef HAVE_PK_BF16
typedef __attribute__((ext_vector_type(2))) __bf16 v2bf;
__device__ __forceinline__ unsigned pack2(float a, float b) {
    v2bf r = __builtin_amdgcn_cvt_pk_bf16_f32(a, b);
    return __builtin_bit_cast(unsigned, r);
}
#else
__device__ __forceinline__ unsigned pack2(float a, float b) {
    return (unsigned)f2bfbits(a) | ((unsigned)f2bfbits(b) << 16);
}
#endif

__device__ __forceinline__ v8bf ld8(const unsigned short* p) {
    return *(const v8bf*)p;                         // 16B vector load
}
__device__ __forceinline__ v16bf cat8(v8bf a, v8bf b) {
    return __builtin_shufflevector(a, b, 0,1,2,3,4,5,6,7,8,9,10,11,12,13,14,15);
}
__device__ __forceinline__ v16bf ld16(const unsigned short* p) {
    return cat8(ld8(p), ld8(p + 8));                // 2 x 16B vector loads
}

// ---------------------------------------------------------------- utilities
__global__ void zero_f32(float* __restrict__ p, int n) {
    int i = blockIdx.x * blockDim.x + threadIdx.x;
    if (i < n) p[i] = 0.f;
}
__global__ void copy_f32(const float* __restrict__ s, float* __restrict__ d, int n) {
    int i = blockIdx.x * blockDim.x + threadIdx.x;
    if (i < n) d[i] = s[i];
}
// fp32 -> bf16 (2 elements per thread, packed dword stores)
__global__ void cvt_bf16(const float* __restrict__ in,
                         unsigned short* __restrict__ out, int n2) {
    int i = blockIdx.x * blockDim.x + threadIdx.x;
    if (i < n2) {
        const float2 v = *(const float2*)(in + 2 * (size_t)i);
        *(unsigned*)(out + 2 * (size_t)i) = pack2(v.x, v.y);
    }
}

// ------------------------------------------------- edge scatter (segment_sum)
__global__ void scatter_edges(const int* __restrict__ ei,
                              const float* __restrict__ h,
                              float* __restrict__ S, float* __restrict__ deg) {
    int idx = blockIdx.x * blockDim.x + threadIdx.x;
    const int chunks = DIM / 4;
    const int total  = NE * chunks;
    if (idx < total) {
        int e = idx / chunks;
        int c = (idx - e * chunks) * 4;
        int dst = ei[e], src = ei[NE + e];
        const float4 v = *(const float4*)(h + (size_t)src * DIM + c);
        atomicAdd(&S[(size_t)dst * DIM + c + 0], v.x);
        atomicAdd(&S[(size_t)dst * DIM + c + 1], v.y);
        atomicAdd(&S[(size_t)dst * DIM + c + 2], v.z);
        atomicAdd(&S[(size_t)dst * DIM + c + 3], v.w);
    }
    if (idx < NE) atomicAdd(&deg[ei[idx]], 1.0f);
}

// --------------------------------------------------------- WMMA GEMM kernel
// Y = X[MxK](bf16) @ W[NoutxK](bf16)^T + epilogue.  64x64 block tile, 4 waves,
// all fragments direct aligned 16B global loads (weights are L2-resident).
// mode 0: +bias ; mode 1: +deg[row]*bias ; mode 2: relu(+bias+res)
__global__ __launch_bounds__(128)
void gemm_wmma(const unsigned short* __restrict__ X,
               const unsigned short* __restrict__ W,
               const float* __restrict__ bias, const float* __restrict__ deg,
               const float* __restrict__ res,
               float* __restrict__ outF, unsigned short* __restrict__ outBF,
               int Nout, int K, int mode) {
    const int tid  = threadIdx.x;
    const int lane = tid & 31;
    const int wv   = tid >> 5;
    const int n0   = blockIdx.x * 64;
    const int m0   = blockIdx.y * 64;
    const int l15  = lane & 15;
    const int hiA  = (lane >= 16) ? 8 : 0;
    const int hi16 = (lane >= 16) ? 16 : 0;
    const int off8 = (lane >= 16) ? 8 : 0;

    // A: per-lane row, K 0..15 (lo) / 16..31 (hi)
    const unsigned short* xp  = X + (size_t)(m0 + wv * 16 + l15) * K + hi16;
    const unsigned short* wp0 = W + (size_t)(n0 +  0 + l15) * K + hi16;
    const unsigned short* wp1 = W + (size_t)(n0 + 16 + l15) * K + hi16;
    const unsigned short* wp2 = W + (size_t)(n0 + 32 + l15) * K + hi16;
    const unsigned short* wp3 = W + (size_t)(n0 + 48 + l15) * K + hi16;

    v8f acc[4] = {};
    for (int k0 = 0; k0 < K; k0 += 32) {
        const v16bf a = ld16(xp + k0);
        acc[0] = __builtin_amdgcn_wmma_f32_16x16x32_bf16(
            false, a, false, ld16(wp0 + k0), (short)0, acc[0], false, false);
        acc[1] = __builtin_amdgcn_wmma_f32_16x16x32_bf16(
            false, a, false, ld16(wp1 + k0), (short)0, acc[1], false, false);
        acc[2] = __builtin_amdgcn_wmma_f32_16x16x32_bf16(
            false, a, false, ld16(wp2 + k0), (short)0, acc[2], false, false);
        acc[3] = __builtin_amdgcn_wmma_f32_16x16x32_bf16(
            false, a, false, ld16(wp3 + k0), (short)0, acc[3], false, false);
    }
#pragma unroll
    for (int nf = 0; nf < 4; ++nf) {
#pragma unroll
        for (int j = 0; j < 8; ++j) {
            int row = m0 + wv * 16 + j + hiA;
            int col = n0 + nf * 16 + l15;
            float v = acc[nf][j];
            if (mode == 1) v += deg[row] * bias[col];
            else           v += bias[col];
            if (mode == 2) { v += res[(size_t)row * Nout + col]; v = fmaxf(v, 0.f); }
            if (outF)  outF [(size_t)row * Nout + col] = v;
            if (outBF) outBF[(size_t)row * Nout + col] = f2bfbits(v);
        }
    }
}

// ------------------------------------------- V transpose: Vt[h][d][node] bf16
__global__ __launch_bounds__(256)
void transpose_v(const unsigned short* __restrict__ qkv,
                 unsigned short* __restrict__ Vt) {
    __shared__ unsigned short T[DH][66];
    const int tid = threadIdx.x;
    const int n0  = blockIdx.x * 64;
    const int hh  = blockIdx.y;
    const unsigned short* base = qkv + (size_t)n0 * (3 * DIM) + 2 * DIM + hh * DH;
    for (int i = tid; i < 64 * 24; i += 256) {       // 24 dwords per 48-short row
        int r = i / 24, dw = i % 24;
        unsigned v = *(const unsigned*)(base + (size_t)r * (3 * DIM) + 2 * dw);
        T[2 * dw    ][r] = (unsigned short)(v & 0xFFFFu);
        T[2 * dw + 1][r] = (unsigned short)(v >> 16);
    }
    __syncthreads();
    for (int i = tid; i < DH * 32; i += 256) {       // write 2 nodes per dword
        int d = i / 32, p = i % 32;
        unsigned v = (unsigned)T[d][2 * p] | ((unsigned)T[d][2 * p + 1] << 16);
        *(unsigned*)(Vt + (size_t)(hh * DH + d) * N_NODES + n0 + 2 * p) = v;
    }
}

// ----------------------------------------------------- flash attention (MHA)
// One wave per (head, 16 query rows); key tiles of 32; online softmax.
// Emits O directly as bf16 (input of the next WMMA GEMM).
#define PROW 40
__global__ __launch_bounds__(32)
void flash_attn(const unsigned short* __restrict__ qkv,
                const unsigned short* __restrict__ Vt,
                unsigned short* __restrict__ Obf) {
    __shared__ unsigned short Pp[16 * PROW];         // P tile, [row][key] in natural key order
    const int lane = threadIdx.x;
    const int hh   = blockIdx.y;
    const int qb   = blockIdx.x * 16;
    const int l15  = lane & 15;
    const int hi8  = (lane >= 16) ? 8 : 0;
    const int hi16 = (lane >= 16) ? 16 : 0;
    const int off8 = (lane >= 16) ? 8 : 0;
    const float sc = 0.14433756729740643f;           // 1/sqrt(48)
    const int LD   = 3 * DIM;

    v8bf z8;
#pragma unroll
    for (int e = 0; e < 8; ++e) z8[e] = bits2bf(0);
    const v16bf z16 = cat8(z8, z8);

    // Q fragments (16x64, DH padded 48->64 with zeros), A layout
    const unsigned short* qrow = qkv + (size_t)(qb + l15) * LD + hh * DH;
    const v16bf aq0 = ld16(qrow + hi16); // K 0..31
    const v16bf aq1 = (lane < 16) ? ld16(qrow + 32) : z16;          // K 32..63

    float m[8], ssum[8];
    v8f oacc[3] = {};
#pragma unroll
    for (int j = 0; j < 8; ++j) { m[j] = -3.0e38f; ssum[j] = 0.f; }

    const int pos0 = l15;                            // key l15
    const int pos1 = 16 + l15;                       // key 16+l15

    for (int kb = 0; kb < N_NODES; kb += 32) {
        // ---- scores: two 16x16 C tiles over key groups of 16, K-inner = 64
        v8f c[2] = {};
#pragma unroll
        for (int g = 0; g < 2; ++g) {
            const unsigned short* kr =
                qkv + (size_t)(kb + g * 16 + l15) * LD + DIM + hh * DH;
            const v16bf b0 = ld16(kr + hi16);                    // kk 0..31
            v16bf b1 = (lane < 16) ? ld16(kr + 32) : z16;        // kk 32..47 | pad
            c[g] = __builtin_amdgcn_wmma_f32_16x16x32_bf16(
                false, aq0, false, b0, (short)0, c[g], false, false);
            c[g] = __builtin_amdgcn_wmma_f32_16x16x32_bf16(
                false, aq1, false, b1, (short)0, c[g], false, false);
        }
        // ---- online softmax (row = j + hi8; 16-lane butterfly reductions)
#pragma unroll
        for (int j = 0; j < 8; ++j) {
            float v0 = c[0][j] * sc, v1 = c[1][j] * sc;
            float rmax = fmaxf(v0, v1);
            rmax = fmaxf(rmax, __shfl_xor(rmax, 1));
            rmax = fmaxf(rmax, __shfl_xor(rmax, 2));
            rmax = fmaxf(rmax, __shfl_xor(rmax, 4));
            rmax = fmaxf(rmax, __shfl_xor(rmax, 8));
            float mn    = fmaxf(m[j], rmax);
            float alpha = __expf(m[j] - mn);
            float p0 = __expf(v0 - mn);
            float p1 = __expf(v1 - mn);
            float rs = p0 + p1;
            rs += __shfl_xor(rs, 1);
            rs += __shfl_xor(rs, 2);
            rs += __shfl_xor(rs, 4);
            rs += __shfl_xor(rs, 8);
            ssum[j] = ssum[j] * alpha + rs;
            m[j]    = mn;
#pragma unroll
            for (int t = 0; t < 3; ++t) oacc[t][j] *= alpha;
            Pp[(j + hi8) * PROW + pos0] = f2bfbits(p0);
            Pp[(j + hi8) * PROW + pos1] = f2bfbits(p1);
        }
        __syncthreads();
        // ---- O += P(16x32) @ V(32x48), V fragments direct from Vt (global)
        const v16bf ap = ld16(&Pp[l15 * PROW + hi16]);
#pragma unroll
        for (int t = 0; t < 3; ++t) {
            const v16bf bv = ld16(
                Vt + (size_t)(hh * DH + t * 16 + l15) * N_NODES + kb + hi16);
            oacc[t] = __builtin_amdgcn_wmma_f32_16x16x32_bf16(
                false, ap, false, bv, (short)0, oacc[t], false, false);
        }
        __syncthreads();
    }
#pragma unroll
    for (int t = 0; t < 3; ++t) {
#pragma unroll
        for (int j = 0; j < 8; ++j) {
            int row = qb + j + hi8;
            int col = hh * DH + t * 16 + l15;
            Obf[(size_t)row * DIM + col] = f2bfbits(oacc[t][j] / ssum[j]);
        }
    }
}

// ---------------------------------------------------- final sigmoid(h@Wo^T+b)
__global__ __launch_bounds__(128)
void out_proj(const float* __restrict__ h, const float* __restrict__ Wo,
              const float* __restrict__ bo, float* __restrict__ out) {
    int node = blockIdx.x * 4 + (threadIdx.x >> 5);
    int lane = threadIdx.x & 31;
    float s = 0.f;
    for (int d = lane; d < DIM; d += 32) s += h[(size_t)node * DIM + d] * Wo[d];
    s += __shfl_xor(s, 16);
    s += __shfl_xor(s, 8);
    s += __shfl_xor(s, 4);
    s += __shfl_xor(s, 2);
    s += __shfl_xor(s, 1);
    if (lane == 0) out[node] = 1.f / (1.f + __expf(-(s + bo[0])));
}

// --------------------------------------------------------------------- host
extern "C" void kernel_launch(void* const* d_in, const int* in_sizes, int n_in,
                              void* d_out, int out_size, void* d_ws, size_t ws_size,
                              hipStream_t stream) {
    const float* emb = (const float*)d_in[0];
    const int*   ei  = (const int*)  d_in[1];
    const float* Wc  = (const float*)d_in[2];
    const float* bc  = (const float*)d_in[3];
    const float* Win = (const float*)d_in[4];
    const float* bin = (const float*)d_in[5];
    const float* Wao = (const float*)d_in[6];
    const float* bao = (const float*)d_in[7];
    const float* Wo  = (const float*)d_in[8];
    const float* bo  = (const float*)d_in[9];
    (void)in_sizes; (void)n_in; (void)out_size; (void)ws_size;

    char* ws = (char*)d_ws;
    size_t off = 0;
    auto alloc = [&](size_t bytes) -> void* {
        void* p = ws + off;
        off = (off + bytes + 255) & ~(size_t)255;
        return p;
    };
    float* S    = (float*)alloc((size_t)N_NODES * DIM * 4);
    float* deg  = (float*)alloc((size_t)N_NODES * 4);
    float* h    = (float*)alloc((size_t)N_NODES * DIM * 4);
    unsigned short* Sbf    = (unsigned short*)alloc((size_t)N_NODES * DIM * 2);
    unsigned short* hnewbf = (unsigned short*)alloc((size_t)N_NODES * DIM * 2);
    unsigned short* attObf = (unsigned short*)alloc((size_t)N_NODES * DIM * 2);
    unsigned short* qkvb   = (unsigned short*)alloc((size_t)N_NODES * 3 * DIM * 2);
    unsigned short* Vt     = (unsigned short*)alloc((size_t)N_NODES * DIM * 2);
    unsigned short* Wcbf   = (unsigned short*)alloc((size_t)NL * DIM * DIM * 2);
    unsigned short* Winbf  = (unsigned short*)alloc((size_t)3 * DIM * DIM * 2);
    unsigned short* Waobf  = (unsigned short*)alloc((size_t)DIM * DIM * 2);

    const int ND = N_NODES * DIM;

    // one-time weight conversions (bf16)
    {
        int n2 = NL * DIM * DIM / 2;
        cvt_bf16<<<(n2 + 255) / 256, 256, 0, stream>>>(Wc, Wcbf, n2);
        n2 = 3 * DIM * DIM / 2;
        cvt_bf16<<<(n2 + 255) / 256, 256, 0, stream>>>(Win, Winbf, n2);
        n2 = DIM * DIM / 2;
        cvt_bf16<<<(n2 + 255) / 256, 256, 0, stream>>>(Wao, Waobf, n2);
    }

    copy_f32<<<(ND + 255) / 256, 256, 0, stream>>>(emb, h, ND);

    for (int l = 0; l < NL; ++l) {
        zero_f32<<<(ND + 255) / 256, 256, 0, stream>>>(S, ND);
        zero_f32<<<(N_NODES + 255) / 256, 256, 0, stream>>>(deg, N_NODES);

        const int scat_total = NE * (DIM / 4);
        scatter_edges<<<(scat_total + 255) / 256, 256, 0, stream>>>(ei, h, S, deg);

        cvt_bf16<<<(ND / 2 + 255) / 256, 256, 0, stream>>>(S, Sbf, ND / 2);

        // h_new(bf16) = S @ Wc[l]^T + deg*b_conv[l]
        gemm_wmma<<<dim3(DIM / 64, N_NODES / 64), 128, 0, stream>>>(
            Sbf, Wcbf + (size_t)l * DIM * DIM, bc + (size_t)l * DIM, deg, nullptr,
            nullptr, hnewbf, DIM, DIM, 1);

        // qkv (bf16) = h_new @ W_in^T + b_in
        gemm_wmma<<<dim3(3 * DIM / 64, N_NODES / 64), 128, 0, stream>>>(
            hnewbf, Winbf, bin, nullptr, nullptr,
            nullptr, qkvb, 3 * DIM, DIM, 0);

        // V -> Vt[h][d][node] for K-contiguous B fragments
        transpose_v<<<dim3(N_NODES / 64, NH), 256, 0, stream>>>(qkvb, Vt);

        // attO(bf16) = softmax(QK^T/sqrt(DH)) V
        flash_attn<<<dim3(N_NODES / 16, NH), 32, 0, stream>>>(qkvb, Vt, attObf);

        // h = relu(attO @ Wao^T + bao + h)
        gemm_wmma<<<dim3(DIM / 64, N_NODES / 64), 128, 0, stream>>>(
            attObf, Waobf, bao, nullptr, h,
            h, nullptr, DIM, DIM, 2);
    }
    out_proj<<<N_NODES / 4, 128, 0, stream>>>(h, Wo, bo, (float*)d_out);
}
